// EGNN_2370821947746
// MI455X (gfx1250) — matrix operand order repaired and checked by reading, not compiled, not run on hardware
//
#include <hip/hip_runtime.h>

#define NN 50000
#define EE 800000
#define GG 100
#define HH 64
#define LL 4

typedef __attribute__((ext_vector_type(16))) _Float16 v16h;
typedef __attribute__((ext_vector_type(8)))  _Float16 v8h;
typedef __attribute__((ext_vector_type(8)))  float    v8f;

__device__ __forceinline__ float sigm(float x)  { return 1.0f / (1.0f + __expf(-x)); }
__device__ __forceinline__ float siluf(float x) { return x * sigm(x); }

__device__ __forceinline__ v8f zero8f() {
    v8f z;
#pragma unroll
    for (int i = 0; i < 8; ++i) z[i] = 0.0f;
    return z;
}
__device__ __forceinline__ v16h zero16h() {
    v16h z;
#pragma unroll
    for (int i = 0; i < 16; ++i) z[i] = (_Float16)0.0f;
    return z;
}

__device__ __forceinline__ v8f wmma_f16(v16h a, v16h b, v8f c) {
    return __builtin_amdgcn_wmma_f32_16x16x32_f16(false, a, false, b, (short)0, c, false, false);
}

// A fragment (16x32 f16): lane holds row m = lane&15; element e -> K = (e&7)+8g+16*(e>>3).
__device__ __forceinline__ v16h afrag_g(const float* __restrict__ seg, int g, float scale) {
    const float* p = seg + 8 * g;
    float4 x0 = *(const float4*)(p);
    float4 x1 = *(const float4*)(p + 4);
    float4 y0 = *(const float4*)(p + 16);
    float4 y1 = *(const float4*)(p + 20);
    v16h a;
    a[0]  = (_Float16)(x0.x * scale); a[1]  = (_Float16)(x0.y * scale);
    a[2]  = (_Float16)(x0.z * scale); a[3]  = (_Float16)(x0.w * scale);
    a[4]  = (_Float16)(x1.x * scale); a[5]  = (_Float16)(x1.y * scale);
    a[6]  = (_Float16)(x1.z * scale); a[7]  = (_Float16)(x1.w * scale);
    a[8]  = (_Float16)(y0.x * scale); a[9]  = (_Float16)(y0.y * scale);
    a[10] = (_Float16)(y0.z * scale); a[11] = (_Float16)(y0.w * scale);
    a[12] = (_Float16)(y1.x * scale); a[13] = (_Float16)(y1.y * scale);
    a[14] = (_Float16)(y1.z * scale); a[15] = (_Float16)(y1.w * scale);
    return a;
}

// A fragment from LDS staging row (stride 72 halves)
__device__ __forceinline__ v16h afrag_lds(const _Float16* row, int g, int kt2) {
    const _Float16* p = row + 32 * kt2 + 8 * g;
    v8h a0 = *(const v8h*)(p);
    v8h a1 = *(const v8h*)(p + 16);
    return __builtin_shufflevector(a0, a1, 0,1,2,3,4,5,6,7,8,9,10,11,12,13,14,15);
}

// B fragment from LDS weights: lane holds column n; element e -> K = (e&7)+8g+16*(e>>3) + 32kt.
// wt rows are padded to `stride` halves (bank-conflict-free pitches).
__device__ __forceinline__ v16h bfrag_lds(const _Float16* wt, int stride, int ncol, int kt, int g) {
    const _Float16* p = wt + ncol * stride + 32 * kt + 8 * g;
    v8h a0 = *(const v8h*)(p);
    v8h a1 = *(const v8h*)(p + 16);
    return __builtin_shufflevector(a0, a1, 0,1,2,3,4,5,6,7,8,9,10,11,12,13,14,15);
}

__device__ __forceinline__ float redux16(float v) {
    v += __shfl_xor(v, 1, 32);
    v += __shfl_xor(v, 2, 32);
    v += __shfl_xor(v, 4, 32);
    v += __shfl_xor(v, 8, 32);
    return v;
}

// Block-cooperative copy of [64][Kpad] f16 weights into LDS with padded row stride.
__device__ __forceinline__ void copy_w_lds(_Float16* dst, const _Float16* __restrict__ src,
                                           int Kpad, int stride) {
    const int chunks = 64 * Kpad / 8;
    const int cpr = Kpad / 8;
    for (int c = threadIdx.x; c < chunks; c += blockDim.x) {
        int row = c / cpr;
        int ko  = (c - row * cpr) * 8;
        *(v8h*)(dst + row * stride + ko) = *(const v8h*)(src + (size_t)row * Kpad + ko);
    }
}

// ---------------- weight transpose + f16 conversion ----------------
__global__ void wprep_kernel(const float* __restrict__ src, _Float16* __restrict__ dst,
                             int Ksrc, int Kpad) {
    int i = blockIdx.x * blockDim.x + threadIdx.x;
    int tot = LL * 64 * Kpad;
    if (i >= tot) return;
    int k = i % Kpad;
    int n = (i / Kpad) % 64;
    int l = i / (Kpad * 64);
    float v = (k < Ksrc) ? src[((size_t)l * Ksrc + k) * 64 + n] : 0.0f;
    dst[((size_t)l * 64 + n) * Kpad + k] = (_Float16)v;
}

__global__ void zero_kernel(float* __restrict__ p, int n) {
    int i = blockIdx.x * blockDim.x + threadIdx.x;
    if (i < n) p[i] = 0.0f;
}

__global__ void embed_kernel(const float* __restrict__ na, const float* __restrict__ tg,
                             const int* __restrict__ batch, const float* __restrict__ emb_w,
                             const float* __restrict__ emb_b, const float* __restrict__ positions,
                             float* __restrict__ h, float* __restrict__ pos) {
    int i = blockIdx.x * blockDim.x + threadIdx.x;
    if (i >= NN * 64) return;
    int n = i >> 6;
    int f = i & 63;
    float acc = emb_b[f];
#pragma unroll
    for (int k = 0; k < 7; ++k)
        acc += (na[n * 7 + k] * 0.25f) * emb_w[k * 64 + f];
    acc += tg[batch[n]] * emb_w[7 * 64 + f];
    h[i] = acc;
    if (f < 3) pos[n * 3 + f] = positions[n * 3 + f];
}

__global__ void len0_kernel(const float* __restrict__ positions, const float* __restrict__ shifts,
                            const int* __restrict__ ei, float* __restrict__ len0) {
    int e = blockIdx.x * blockDim.x + threadIdx.x;
    if (e >= EE) return;
    int s = ei[e], r = ei[EE + e];
    float vx = positions[r * 3 + 0] - positions[s * 3 + 0] + shifts[e * 3 + 0];
    float vy = positions[r * 3 + 1] - positions[s * 3 + 1] + shifts[e * 3 + 1];
    float vz = positions[r * 3 + 2] - positions[s * 3 + 2] + shifts[e * 3 + 2];
    len0[e] = sqrtf(vx * vx + vy * vy + vz * vz);
}

// ---------------- edge message kernel (WMMA, edge MLP + gate + scatter) ----------------
__global__ __launch_bounds__(256) void egnn_edge_msg(
    const float* __restrict__ h, const float* __restrict__ pos,
    const float* __restrict__ shifts, const int* __restrict__ ei,
    const float* __restrict__ len0,
    const _Float16* __restrict__ w1g, const float* __restrict__ b1,
    const _Float16* __restrict__ w2g, const float* __restrict__ b2,
    const float* __restrict__ einf,
    float* __restrict__ message) {
    __shared__ __align__(16) _Float16 w1s[64 * 168];        // 21504 B (pitch 336B: conflict-free)
    __shared__ __align__(16) _Float16 w2s[64 * 72];         //  9216 B (pitch 144B)
    __shared__ __align__(16) _Float16 stage[8][16][72];     // 18432 B
    copy_w_lds(w1s, w1g, 160, 168);
    copy_w_lds(w2s, w2g, 64, 72);
    __syncthreads();

    const int lane = threadIdx.x & 31;
    const int wv   = threadIdx.x >> 5;
    const int m16  = lane & 15;
    const int g    = lane >> 4;

    float b1v[4], b2v[4], einfv[4];
#pragma unroll
    for (int nt = 0; nt < 4; ++nt) {
        int f = 16 * nt + m16;
        b1v[nt] = b1[f]; b2v[nt] = b2[f]; einfv[nt] = einf[f];
    }

    const int T  = (EE + 15) >> 4;
    const int gw = blockIdx.x * 8 + wv;
    const int nw = gridDim.x * 8;
    const int iters = (T + nw - 1) / nw;

    for (int it = 0; it < iters; ++it) {
        asm volatile("" ::: "memory");  // keep fragment loads inside the loop (no LICM spill)
        int t = gw + it * nw;
        const bool tv = (t < T);
        if (!tv) t = 0;
        const int base = t << 4;
        int eL = base + m16;
        const bool ev = tv && (eL < EE);
        if (eL >= EE) eL = EE - 1;

        const int snd = ei[eL];
        const int rcv = ei[EE + eL];
        float vx = pos[rcv * 3 + 0] - pos[snd * 3 + 0] + shifts[eL * 3 + 0];
        float vy = pos[rcv * 3 + 1] - pos[snd * 3 + 1] + shifts[eL * 3 + 1];
        float vz = pos[rcv * 3 + 2] - pos[snd * 3 + 2] + shifts[eL * 3 + 2];
        const float len = sqrtf(vx * vx + vy * vy + vz * vz);
        const float l0  = len0[eL];

        v8f acc[4];
#pragma unroll
        for (int nt = 0; nt < 4; ++nt) acc[nt] = zero8f();

        const float* hs = h + (size_t)snd * 64;
        const float* hr = h + (size_t)rcv * 64;
#pragma unroll
        for (int kt = 0; kt < 4; ++kt) {
            const float* seg = (kt < 2) ? (hs + 32 * kt) : (hr + 32 * (kt - 2));
            v16h a = afrag_g(seg, g, 1.0f);
#pragma unroll
            for (int nt = 0; nt < 4; ++nt)
                acc[nt] = wmma_f16(a, bfrag_lds(w1s, 168, 16 * nt + m16, kt, g), acc[nt]);
        }
        {   // kt = 4: K=128 length, K=129 length_0
            v16h a = zero16h();
            a[0] = (_Float16)((g == 0) ? len : 0.0f);
            a[1] = (_Float16)((g == 0) ? l0  : 0.0f);
#pragma unroll
            for (int nt = 0; nt < 4; ++nt)
                acc[nt] = wmma_f16(a, bfrag_lds(w1s, 168, 16 * nt + m16, 4, g), acc[nt]);
        }

#pragma unroll
        for (int nt = 0; nt < 4; ++nt)
#pragma unroll
            for (int r = 0; r < 8; ++r)
                stage[wv][r + 8 * g][16 * nt + m16] = (_Float16)siluf(acc[nt][r] + b1v[nt]);
        __syncthreads();

        v8f c2[4];
#pragma unroll
        for (int nt = 0; nt < 4; ++nt) c2[nt] = zero8f();
#pragma unroll
        for (int kt2 = 0; kt2 < 2; ++kt2) {
            v16h a = afrag_lds(&stage[wv][m16][0], g, kt2);
#pragma unroll
            for (int nt = 0; nt < 4; ++nt)
                c2[nt] = wmma_f16(a, bfrag_lds(w2s, 72, 16 * nt + m16, kt2, g), c2[nt]);
        }
        __syncthreads();

        float part[8];
#pragma unroll
        for (int r = 0; r < 8; ++r) part[r] = 0.0f;
#pragma unroll
        for (int nt = 0; nt < 4; ++nt)
#pragma unroll
            for (int r = 0; r < 8; ++r) {
                float m_ = siluf(c2[nt][r] + b2v[nt]);
                c2[nt][r] = m_;
                part[r] += m_ * einfv[nt];
            }
        float scv[8];
#pragma unroll
        for (int r = 0; r < 8; ++r) scv[r] = sigm(redux16(part[r]));

        int rv_enc = ev ? rcv : -1;
#pragma unroll
        for (int r = 0; r < 8; ++r) {
            int rvm = __shfl(rv_enc, r + 8 * g, 32);
            if (rvm >= 0) {
                float s = scv[r];
#pragma unroll
                for (int nt = 0; nt < 4; ++nt)
                    atomicAdd(&message[(size_t)rvm * 64 + 16 * nt + m16], s * c2[nt][r]);
            }
        }
    }
}

// ---------------- edge position kernel (WMMA, pos MLP + vector scatter) ----------------
__global__ __launch_bounds__(256) void egnn_edge_pos(
    const float* __restrict__ h, const float* __restrict__ pos,
    const float* __restrict__ shifts, const int* __restrict__ ei,
    const float* __restrict__ len0,
    const _Float16* __restrict__ w1g, const float* __restrict__ b1,
    const _Float16* __restrict__ w2g, const float* __restrict__ b2,
    const float* __restrict__ w3,
    float* __restrict__ dpos) {
    __shared__ __align__(16) _Float16 w1s[64 * 168];
    __shared__ __align__(16) _Float16 w2s[64 * 72];
    __shared__ __align__(16) _Float16 stage[8][16][72];
    copy_w_lds(w1s, w1g, 160, 168);
    copy_w_lds(w2s, w2g, 64, 72);
    __syncthreads();

    const int lane = threadIdx.x & 31;
    const int wv   = threadIdx.x >> 5;
    const int m16  = lane & 15;
    const int g    = lane >> 4;

    float b1v[4], b2v[4], w3v[4];
#pragma unroll
    for (int nt = 0; nt < 4; ++nt) {
        int f = 16 * nt + m16;
        b1v[nt] = b1[f]; b2v[nt] = b2[f]; w3v[nt] = w3[f];
    }

    const int T  = (EE + 15) >> 4;
    const int gw = blockIdx.x * 8 + wv;
    const int nw = gridDim.x * 8;
    const int iters = (T + nw - 1) / nw;

    for (int it = 0; it < iters; ++it) {
        asm volatile("" ::: "memory");
        int t = gw + it * nw;
        const bool tv = (t < T);
        if (!tv) t = 0;
        const int base = t << 4;
        int eL = base + m16;
        const bool ev = tv && (eL < EE);
        if (eL >= EE) eL = EE - 1;

        const int snd = ei[eL];
        const int rcv = ei[EE + eL];
        float vx = pos[rcv * 3 + 0] - pos[snd * 3 + 0] + shifts[eL * 3 + 0];
        float vy = pos[rcv * 3 + 1] - pos[snd * 3 + 1] + shifts[eL * 3 + 1];
        float vz = pos[rcv * 3 + 2] - pos[snd * 3 + 2] + shifts[eL * 3 + 2];
        const float len = sqrtf(vx * vx + vy * vy + vz * vz);
        const float l0  = len0[eL];

        v8f acc[4];
#pragma unroll
        for (int nt = 0; nt < 4; ++nt) acc[nt] = zero8f();

        const float* hs = h + (size_t)snd * 64;
        const float* hr = h + (size_t)rcv * 64;
#pragma unroll
        for (int kt = 0; kt < 4; ++kt) {
            const float* seg = (kt < 2) ? (hs + 32 * kt) : (hr + 32 * (kt - 2));
            v16h a = afrag_g(seg, g, 1.0f);
#pragma unroll
            for (int nt = 0; nt < 4; ++nt)
                acc[nt] = wmma_f16(a, bfrag_lds(w1s, 168, 16 * nt + m16, kt, g), acc[nt]);
        }
        {
            v16h a = zero16h();
            a[0] = (_Float16)((g == 0) ? len : 0.0f);
            a[1] = (_Float16)((g == 0) ? l0  : 0.0f);
#pragma unroll
            for (int nt = 0; nt < 4; ++nt)
                acc[nt] = wmma_f16(a, bfrag_lds(w1s, 168, 16 * nt + m16, 4, g), acc[nt]);
        }

#pragma unroll
        for (int nt = 0; nt < 4; ++nt)
#pragma unroll
            for (int r = 0; r < 8; ++r)
                stage[wv][r + 8 * g][16 * nt + m16] = (_Float16)siluf(acc[nt][r] + b1v[nt]);
        __syncthreads();

        v8f c2[4];
#pragma unroll
        for (int nt = 0; nt < 4; ++nt) c2[nt] = zero8f();
#pragma unroll
        for (int kt2 = 0; kt2 < 2; ++kt2) {
            v16h a = afrag_lds(&stage[wv][m16][0], g, kt2);
#pragma unroll
            for (int nt = 0; nt < 4; ++nt)
                c2[nt] = wmma_f16(a, bfrag_lds(w2s, 72, 16 * nt + m16, kt2, g), c2[nt]);
        }
        __syncthreads();

        float pp[8];
#pragma unroll
        for (int r = 0; r < 8; ++r) pp[r] = 0.0f;
#pragma unroll
        for (int nt = 0; nt < 4; ++nt)
#pragma unroll
            for (int r = 0; r < 8; ++r)
                pp[r] += siluf(c2[nt][r] + b2v[nt]) * w3v[nt];
#pragma unroll
        for (int r = 0; r < 8; ++r) pp[r] = redux16(pp[r]);

        int rv_enc = ev ? rcv : -1;
#pragma unroll
        for (int r = 0; r < 8; ++r) {
            int src = r + 8 * g;
            int rvm   = __shfl(rv_enc, src, 32);
            float vxm = __shfl(vx, src, 32);
            float vym = __shfl(vy, src, 32);
            float vzm = __shfl(vz, src, 32);
            if (m16 == 0 && rvm >= 0) {
                atomicAdd(&dpos[(size_t)rvm * 3 + 0], vxm * pp[r]);
                atomicAdd(&dpos[(size_t)rvm * 3 + 1], vym * pp[r]);
                atomicAdd(&dpos[(size_t)rvm * 3 + 2], vzm * pp[r]);
            }
        }
    }
}

// ---------------- node kernel (WMMA) ----------------
__global__ __launch_bounds__(256) void egnn_node(
    float* __restrict__ h, const float* __restrict__ message,
    const _Float16* __restrict__ w1g, const float* __restrict__ b1,
    const _Float16* __restrict__ w2g, const float* __restrict__ b2) {
    __shared__ __align__(16) _Float16 w1s[64 * 136];        // 17408 B (pitch 272B)
    __shared__ __align__(16) _Float16 w2s[64 * 72];
    __shared__ __align__(16) _Float16 stage[8][16][72];
    copy_w_lds(w1s, w1g, 128, 136);
    copy_w_lds(w2s, w2g, 64, 72);
    __syncthreads();

    const int lane = threadIdx.x & 31;
    const int wv   = threadIdx.x >> 5;
    const int m16  = lane & 15;
    const int g    = lane >> 4;

    float b1v[4], b2v[4];
#pragma unroll
    for (int nt = 0; nt < 4; ++nt) {
        int f = 16 * nt + m16;
        b1v[nt] = b1[f];
        b2v[nt] = b2[f];
    }

    const int T  = (NN + 15) >> 4;
    const int gw = blockIdx.x * 8 + wv;
    const int nw = gridDim.x * 8;
    const int iters = (T + nw - 1) / nw;

    for (int it = 0; it < iters; ++it) {
        asm volatile("" ::: "memory");
        int t = gw + it * nw;
        const bool tv = (t < T);
        if (!tv) t = 0;
        const int base = t << 4;
        int nL = base + m16;
        if (nL >= NN) nL = NN - 1;

        const float* hp = h + (size_t)nL * 64;
        const float* mp = message + (size_t)nL * 64;

        v8f acc[4];
#pragma unroll
        for (int nt = 0; nt < 4; ++nt) acc[nt] = zero8f();
#pragma unroll
        for (int kt = 0; kt < 4; ++kt) {
            const float* seg = (kt < 2) ? (hp + 32 * kt) : (mp + 32 * (kt - 2));
            const float scale = (kt < 2) ? 1.0f : 0.0625f;  // message / AVG_NEIGH
            v16h a = afrag_g(seg, g, scale);
#pragma unroll
            for (int nt = 0; nt < 4; ++nt)
                acc[nt] = wmma_f16(a, bfrag_lds(w1s, 136, 16 * nt + m16, kt, g), acc[nt]);
        }
#pragma unroll
        for (int nt = 0; nt < 4; ++nt)
#pragma unroll
            for (int r = 0; r < 8; ++r)
                stage[wv][r + 8 * g][16 * nt + m16] = (_Float16)siluf(acc[nt][r] + b1v[nt]);
        __syncthreads();

        v8f c2[4];
#pragma unroll
        for (int nt = 0; nt < 4; ++nt) c2[nt] = zero8f();
#pragma unroll
        for (int kt2 = 0; kt2 < 2; ++kt2) {
            v16h a = afrag_lds(&stage[wv][m16][0], g, kt2);
#pragma unroll
            for (int nt = 0; nt < 4; ++nt)
                c2[nt] = wmma_f16(a, bfrag_lds(w2s, 72, 16 * nt + m16, kt2, g), c2[nt]);
        }
        __syncthreads();

#pragma unroll
        for (int nt = 0; nt < 4; ++nt)
#pragma unroll
            for (int r = 0; r < 8; ++r) {
                int nm = base + r + 8 * g;
                if (tv && nm < NN) {
                    size_t idx = (size_t)nm * 64 + 16 * nt + m16;
                    h[idx] = h[idx] + c2[nt][r] + b2v[nt];
                }
            }
    }
}

__global__ void posupd_kernel(float* __restrict__ pos, const float* __restrict__ dpos) {
    int i = blockIdx.x * blockDim.x + threadIdx.x;
    if (i < NN * 3) pos[i] += dpos[i] * 0.0625f;
}

__global__ void graph_accum(const float* __restrict__ pos, const int* __restrict__ batch,
                            float* __restrict__ gs, float* __restrict__ gc) {
    int n = blockIdx.x * blockDim.x + threadIdx.x;
    if (n >= NN) return;
    int b = batch[n];
    atomicAdd(&gs[b * 3 + 0], pos[n * 3 + 0]);
    atomicAdd(&gs[b * 3 + 1], pos[n * 3 + 1]);
    atomicAdd(&gs[b * 3 + 2], pos[n * 3 + 2]);
    atomicAdd(&gc[b], 1.0f);
}

__global__ void final_kernel(const float* __restrict__ h, const float* __restrict__ out_w,
                             const float* __restrict__ out_b, const float* __restrict__ pos,
                             const float* __restrict__ positions, const int* __restrict__ batch,
                             const float* __restrict__ gs, const float* __restrict__ gc,
                             float* __restrict__ out) {
    int i = blockIdx.x * blockDim.x + threadIdx.x;
    if (i >= NN * 10) return;
    int n = i / 10;
    int j = i % 10;
    if (j < 7) {
        float acc = out_b[j];
        const float* hr = h + (size_t)n * 64;
#pragma unroll 8
        for (int k = 0; k < 64; ++k) acc += hr[k] * out_w[k * 8 + j];
        out[i] = acc;
    } else {
        int c = j - 7;
        int b = batch[n];
        float cnt  = fmaxf(gc[b], 1.0f);
        float mean = gs[b * 3 + c] / cnt;
        out[i] = pos[n * 3 + c] - mean - positions[n * 3 + c];
    }
}

extern "C" void kernel_launch(void* const* d_in, const int* in_sizes, int n_in,
                              void* d_out, int out_size, void* d_ws, size_t ws_size,
                              hipStream_t stream) {
    (void)in_sizes; (void)n_in; (void)out_size; (void)ws_size;
    const float* node_attrs = (const float*)d_in[0];
    const float* positions  = (const float*)d_in[1];
    const float* t_g        = (const float*)d_in[2];
    const float* shifts     = (const float*)d_in[3];
    const int*   edge_index = (const int*)d_in[4];
    const int*   batch      = (const int*)d_in[5];
    const float* emb_w      = (const float*)d_in[6];
    const float* emb_b      = (const float*)d_in[7];
    const float* out_w      = (const float*)d_in[8];
    const float* out_b      = (const float*)d_in[9];
    const float* edge_w1    = (const float*)d_in[10];
    const float* edge_b1    = (const float*)d_in[11];
    const float* edge_w2    = (const float*)d_in[12];
    const float* edge_b2    = (const float*)d_in[13];
    const float* einf_w     = (const float*)d_in[14];
    const float* node_w1    = (const float*)d_in[15];
    const float* node_b1    = (const float*)d_in[16];
    const float* node_w2    = (const float*)d_in[17];
    const float* node_b2    = (const float*)d_in[18];
    const float* pos_w1     = (const float*)d_in[19];
    const float* pos_b1     = (const float*)d_in[20];
    const float* pos_w2     = (const float*)d_in[21];
    const float* pos_b2     = (const float*)d_in[22];
    const float* pos_w3     = (const float*)d_in[23];
    float* out = (float*)d_out;

    char* wsb = (char*)d_ws;
    size_t off = 0;
    auto alloc = [&](size_t bytes) -> char* {
        char* p = wsb + off;
        off = (off + bytes + 255) & ~(size_t)255;
        return p;
    };
    float* h       = (float*)alloc((size_t)NN * 64 * 4);
    float* pos     = (float*)alloc((size_t)NN * 3 * 4);
    float* message = (float*)alloc((size_t)NN * 64 * 4);
    float* dpos    = (float*)alloc((size_t)NN * 3 * 4);
    float* len0    = (float*)alloc((size_t)EE * 4);
    float* gsums   = (float*)alloc((size_t)GG * 4 * 4);  // [G*3] sums + [G] counts
    _Float16* ew1t = (_Float16*)alloc((size_t)LL * 64 * 160 * 2);
    _Float16* pw1t = (_Float16*)alloc((size_t)LL * 64 * 160 * 2);
    _Float16* nw1t = (_Float16*)alloc((size_t)LL * 64 * 128 * 2);
    _Float16* ew2t = (_Float16*)alloc((size_t)LL * 64 * 64 * 2);
    _Float16* pw2t = (_Float16*)alloc((size_t)LL * 64 * 64 * 2);
    _Float16* nw2t = (_Float16*)alloc((size_t)LL * 64 * 64 * 2);

    wprep_kernel<<<(LL * 64 * 160 + 255) / 256, 256, 0, stream>>>(edge_w1, ew1t, 130, 160);
    wprep_kernel<<<(LL * 64 * 160 + 255) / 256, 256, 0, stream>>>(pos_w1, pw1t, 130, 160);
    wprep_kernel<<<(LL * 64 * 128 + 255) / 256, 256, 0, stream>>>(node_w1, nw1t, 128, 128);
    wprep_kernel<<<(LL * 64 * 64 + 255) / 256, 256, 0, stream>>>(edge_w2, ew2t, 64, 64);
    wprep_kernel<<<(LL * 64 * 64 + 255) / 256, 256, 0, stream>>>(pos_w2, pw2t, 64, 64);
    wprep_kernel<<<(LL * 64 * 64 + 255) / 256, 256, 0, stream>>>(node_w2, nw2t, 64, 64);

    embed_kernel<<<(NN * 64 + 255) / 256, 256, 0, stream>>>(node_attrs, t_g, batch, emb_w,
                                                            emb_b, positions, h, pos);
    len0_kernel<<<(EE + 255) / 256, 256, 0, stream>>>(positions, shifts, edge_index, len0);

    for (int i = 0; i < LL; ++i) {
        zero_kernel<<<(NN * 64 + 255) / 256, 256, 0, stream>>>(message, NN * 64);
        zero_kernel<<<(NN * 3 + 255) / 256, 256, 0, stream>>>(dpos, NN * 3);
        egnn_edge_msg<<<1024, 256, 0, stream>>>(
            h, pos, shifts, edge_index, len0,
            ew1t + (size_t)i * 64 * 160, edge_b1 + i * 64,
            ew2t + (size_t)i * 64 * 64, edge_b2 + i * 64, einf_w + i * 64,
            message);
        egnn_edge_pos<<<1024, 256, 0, stream>>>(
            h, pos, shifts, edge_index, len0,
            pw1t + (size_t)i * 64 * 160, pos_b1 + i * 64,
            pw2t + (size_t)i * 64 * 64, pos_b2 + i * 64, pos_w3 + i * 64,
            dpos);
        egnn_node<<<256, 256, 0, stream>>>(h, message,
            nw1t + (size_t)i * 64 * 128, node_b1 + i * 64,
            nw2t + (size_t)i * 64 * 64, node_b2 + i * 64);
        posupd_kernel<<<(NN * 3 + 255) / 256, 256, 0, stream>>>(pos, dpos);
    }

    zero_kernel<<<(GG * 4 + 255) / 256, 256, 0, stream>>>(gsums, GG * 4);
    graph_accum<<<(NN + 255) / 256, 256, 0, stream>>>(pos, batch, gsums, gsums + 3 * GG);
    final_kernel<<<(NN * 10 + 255) / 256, 256, 0, stream>>>(h, out_w, out_b, pos, positions,
                                                            batch, gsums, gsums + 3 * GG, out);
}
